// MultiHeadAttention_33543694581834
// MI455X (gfx1250) — matrix operand order repaired and checked by reading, not compiled, not run on hardware
//
#include <hip/hip_runtime.h>
#include <hip/hip_bf16.h>
#include <cstdint>
#include <cstddef>

// ---------------- problem constants ----------------
#define B_   8
#define L_   1024
#define D_   1024
#define NH_  16
#define DK_  64
#define AROW 1032      // padded row stride (elements) for LDS tiles

// ---------------- CDNA5 WMMA types ----------------
typedef __attribute__((ext_vector_type(16))) __bf16   v16bf;
typedef __attribute__((ext_vector_type(8)))  __bf16   v8bf;
typedef __attribute__((ext_vector_type(8)))  float    v8f;
typedef __attribute__((ext_vector_type(8)))  uint16_t v8u16;

__device__ __forceinline__ uint16_t f32_to_bf16_bits(float f) {
  uint32_t u = __builtin_bit_cast(uint32_t, f);
  uint32_t r = u + 0x7FFFu + ((u >> 16) & 1u);   // round-to-nearest-even
  return (uint16_t)(r >> 16);
}

__device__ __forceinline__ v8f wmma_bf16(v16bf a, v16bf b, v8f c) {
  // (neg_a, A, neg_b, B, c_mod, C, reuse_a, reuse_b)
  return __builtin_amdgcn_wmma_f32_16x16x32_bf16(false, a, false, b, (short)0, c,
                                                 false, false);
}

__device__ __forceinline__ v16bf make_v16(v8bf lo, v8bf hi) {
  v16bf a;
#pragma unroll
  for (int i = 0; i < 8; ++i) { a[i] = lo[i]; a[i + 8] = hi[i]; }
  return a;
}

// A-matrix 16x32 bf16 fragment: lane<16 holds K{0..7,16..23}, lane>=16 K{8..15,24..31}
__device__ __forceinline__ v16bf load_a_frag16(const uint16_t* base, int stride, int k0) {
  const int lane = threadIdx.x & 31;
  const uint16_t* p = base + (size_t)(lane & 15) * stride + k0 + ((lane >> 4) << 3);
  v8bf lo = *(const v8bf*)(const void*)p;
  v8bf hi = *(const v8bf*)(const void*)(p + 16);
  return make_v16(lo, hi);
}

// B-matrix 32x16 bf16 fragment from an N-major source: lane<16 holds K{0..7,16..23} of
// column (lane&15), lane>=16 holds K{8..15,24..31}.
__device__ __forceinline__ v16bf load_b_frag16(const uint16_t* base, int stride, int k0) {
  const int lane = threadIdx.x & 31;
  const uint16_t* p = base + (size_t)(lane & 15) * stride + k0 + ((lane >> 4) << 3);
  v8bf lo = *(const v8bf*)(const void*)p;
  v8bf hi = *(const v8bf*)(const void*)(p + 16);
  return make_v16(lo, hi);
}

// A fragment built from f32 data (exp(P) values in LDS), converted to bf16 on the fly
__device__ __forceinline__ v16bf load_a_frag_f32(const float* base, int stride, int k0) {
  const int lane = threadIdx.x & 31;
  const float* p = base + (size_t)(lane & 15) * stride + k0 + ((lane >> 4) << 3);
  v16bf a;
#pragma unroll
  for (int i = 0; i < 8; ++i) {
    a[i]     = __builtin_bit_cast(__bf16, f32_to_bf16_bits(p[i]));
    a[i + 8] = __builtin_bit_cast(__bf16, f32_to_bf16_bits(p[i + 16]));
  }
  return a;
}

// ---------------- kernel 1: weight transpose + fp32->bf16 ----------------
// out[h][c][r] = bf16(in[h][r][c])   (H x R x C input)
__global__ void transpose_cvt_kernel(const float* __restrict__ in,
                                     uint16_t* __restrict__ out,
                                     int H, int R, int C) {
  size_t idx = (size_t)blockIdx.x * blockDim.x + threadIdx.x;
  size_t total = (size_t)H * R * C;
  if (idx >= total) return;
  int c = (int)(idx % C);
  size_t t = idx / C;
  int r = (int)(t % R);
  int h = (int)(t / R);
  out[((size_t)h * C + c) * R + r] = f32_to_bf16_bits(in[idx]);
}

// ---------------- kernel 2: per-head QKV projection GEMM ----------------
// 4 waves per block share a 16-row bf16 A tile in LDS; each wave owns a
// 16-n-tile slice and computes 2 n-tiles per k-pass (one A-frag, two WMMAs).
// mode 0: store row-major [h][B][L][64] (Q, K).  mode 1: store transposed
// [h][B][64][L] (V) -- direct 16B stores from the accumulator layout.
__global__ void __launch_bounds__(128) qkv_gemm_kernel(
    const float* __restrict__ x, const uint16_t* __restrict__ wT,
    uint16_t* __restrict__ out, int mode) {
  extern __shared__ __align__(16) char smem_raw[];
  uint16_t* Ab   = (uint16_t*)smem_raw;       // 16 x AROW bf16 A-stage
  uint16_t* Obuf = Ab + 16 * AROW;            // 4 waves x (16 x 24)
  const int tid  = threadIdx.x;
  const int wave = tid >> 5, wlane = tid & 31;
  const int b  = blockIdx.x >> 6;
  const int l0 = (blockIdx.x & 63) << 4;

  // stage A (fp32 -> bf16) into LDS: 16 rows x 1024
  const float* xrow = x + ((size_t)b * L_ + l0) * D_;
  for (int t = tid; t < 4096; t += 128) {
    int row = t >> 8, c4 = (t & 255) << 2;
    float4 f = *(const float4*)(const void*)(xrow + (size_t)row * D_ + c4);
    uint32_t lo = (uint32_t)f32_to_bf16_bits(f.x) | ((uint32_t)f32_to_bf16_bits(f.y) << 16);
    uint32_t hi = (uint32_t)f32_to_bf16_bits(f.z) | ((uint32_t)f32_to_bf16_bits(f.w) << 16);
    uint32_t* d = (uint32_t*)(void*)(Ab + (size_t)row * AROW + c4);
    d[0] = lo; d[1] = hi;
  }
  __syncthreads();

  const int n = wlane & 15, m0 = (wlane >> 4) << 3;
  uint16_t* Ow = Obuf + wave * 16 * 24;

  auto store_tile = [&](int nt, v8f acc) {
    const int h = nt >> 2, kk0 = (nt & 3) << 4;
    if (mode == 0) {
      // stage through this wave's LDS buffer (in-wave DS ordering suffices)
#pragma unroll
      for (int r = 0; r < 8; ++r)
        Ow[(m0 + r) * 24 + n] = f32_to_bf16_bits(acc[r]);
      int row = wlane >> 1, c8 = (wlane & 1) << 3;
      v8u16 val = *(const v8u16*)(const void*)(Ow + row * 24 + c8);
      uint16_t* dst = out + (((size_t)h * B_ + b) * L_ + l0 + row) * 64 + kk0 + c8;
      *(v8u16*)(void*)dst = val;
    } else {
      v8u16 val;
#pragma unroll
      for (int r = 0; r < 8; ++r) val[r] = f32_to_bf16_bits(acc[r]);
      uint16_t* dst = out + (((size_t)h * B_ + b) * 64 + kk0 + n) * (size_t)1024 + l0 + m0;
      *(v8u16*)(void*)dst = val;
    }
  };

  for (int nt = wave * 16; nt < wave * 16 + 16; nt += 2) {
    v8f acc0 = {}, acc1 = {};
    const uint16_t* w0 = wT + (size_t)(nt * 16) * 1024;
    const uint16_t* w1 = w0 + (size_t)16 * 1024;
    for (int k0 = 0; k0 < 1024; k0 += 32) {
      __builtin_prefetch(w0 + (size_t)(wlane & 15) * 1024 + k0 + 128, 0, 1);
      v16bf a  = load_a_frag16(Ab, AROW, k0);
      v16bf b0 = load_b_frag16(w0, 1024, k0);
      v16bf b1 = load_b_frag16(w1, 1024, k0);
      acc0 = wmma_bf16(a, b0, acc0);
      acc1 = wmma_bf16(a, b1, acc1);
    }
    store_tile(nt, acc0);
    store_tile(nt + 1, acc1);
  }
}

// ---------------- kernel 3: attention (scores + softmax + PV) ----------------
// One wave per (h, b, 16 query rows).
__global__ void __launch_bounds__(32) attention_kernel(
    const uint16_t* __restrict__ Qh, const uint16_t* __restrict__ Kh,
    const uint16_t* __restrict__ Vt, const unsigned char* __restrict__ mask,
    uint16_t* __restrict__ Ob) {
  extern __shared__ __align__(16) char smem_raw[];
  float* S = (float*)smem_raw;                      // 16 x AROW f32
  uint16_t* Obuf = (uint16_t*)(S + 16 * AROW);      // 16 x 24
  const int lane = threadIdx.x;
  const int h  = blockIdx.x >> 9;
  const int b  = (blockIdx.x >> 6) & 7;
  const int q0 = (blockIdx.x & 63) << 4;
  const int n = lane & 15, m0 = (lane >> 4) << 3;

  // --- S = Q * K^T (raw scores into LDS) ---
  const uint16_t* qb = Qh + (((size_t)h * B_ + b) * L_ + q0) * 64;
  v16bf aq0 = load_a_frag16(qb, 64, 0);
  v16bf aq1 = load_a_frag16(qb, 64, 32);
  const uint16_t* kb = Kh + (((size_t)h * B_ + b) * L_) * 64;
  for (int s0 = 0; s0 < L_; s0 += 16) {
    v8f c = {};
    v16bf b0 = load_b_frag16(kb + (size_t)s0 * 64, 64, 0);
    v16bf b1 = load_b_frag16(kb + (size_t)s0 * 64, 64, 32);
    c = wmma_bf16(aq0, b0, c);
    c = wmma_bf16(aq1, b1, c);
#pragma unroll
    for (int r = 0; r < 8; ++r) S[(size_t)(m0 + r) * AROW + s0 + n] = c[r];
  }
  __syncthreads();

  // --- scale + mask + softmax (lane = (row, half); contiguous mask reads) ---
  float fact;
  {
    const int row = lane & 15, cb = (lane >> 4) << 9;
    const unsigned char* mrow =
        mask + ((size_t)b * L_ + q0 + row) * (size_t)L_ + cb;
    float* Srow = S + (size_t)row * AROW + cb;
    const float scale = 0.03125f;                 // 1/sqrt(1024)
    float mmax = -3.4e38f;
    for (int i = 0; i < 512; i += 4) {
      uint32_t mb = *(const uint32_t*)(const void*)(mrow + i);
#pragma unroll
      for (int j = 0; j < 4; ++j) {
        float xv = Srow[i + j] * scale;
        if ((mb >> (8 * j)) & 0xffu) xv = -__builtin_inff();
        Srow[i + j] = xv;
        mmax = fmaxf(mmax, xv);
      }
    }
    mmax = fmaxf(mmax, __shfl_xor(mmax, 16, 32));
    float mm = fmaxf(mmax, -1e30f);               // fully-masked row -> finite max
    float ssum = 0.f;
    for (int i = 0; i < 512; ++i) {
      float e = __expf(Srow[i] - mm);
      Srow[i] = e;
      ssum += e;
    }
    ssum += __shfl_xor(ssum, 16, 32);
    fact = (ssum > 0.f) ? 1.0f / ssum : 0.f;      // NaN(all-masked) -> 0 semantics
  }
  __syncthreads();

  // --- O = softmax(S) * V : all 4 output n-tiles share each converted A-frag ---
  v8f acc0 = {}, acc1 = {}, acc2 = {}, acc3 = {};
  const uint16_t* vbase = Vt + ((size_t)h * B_ + b) * 64 * (size_t)1024;
  for (int s0 = 0; s0 < 1024; s0 += 32) {
    __builtin_prefetch(vbase + (size_t)(lane & 15) * 1024 + s0 + 128, 0, 1);
    v16bf a = load_a_frag_f32(S, AROW, s0);
    acc0 = wmma_bf16(a, load_b_frag16(vbase + (size_t)0  * 16 * 1024, 1024, s0), acc0);
    acc1 = wmma_bf16(a, load_b_frag16(vbase + (size_t)1  * 16 * 1024, 1024, s0), acc1);
    acc2 = wmma_bf16(a, load_b_frag16(vbase + (size_t)2  * 16 * 1024, 1024, s0), acc2);
    acc3 = wmma_bf16(a, load_b_frag16(vbase + (size_t)3  * 16 * 1024, 1024, s0), acc3);
  }
#pragma unroll
  for (int nt = 0; nt < 4; ++nt) {
    v8f acc = (nt == 0) ? acc0 : (nt == 1) ? acc1 : (nt == 2) ? acc2 : acc3;
#pragma unroll
    for (int r = 0; r < 8; ++r) {
      float fr = __shfl(fact, m0 + r, 32);
      Obuf[(m0 + r) * 24 + n] = f32_to_bf16_bits(acc[r] * fr);
    }
    // single-wave workgroup: in-wave DS ordering makes this read-after-write safe
    int row = lane >> 1, c8 = (lane & 1) << 3;
    v8u16 val = *(const v8u16*)(const void*)(Obuf + row * 24 + c8);
    uint16_t* dst =
        Ob + ((size_t)b * L_ + q0 + row) * D_ + h * 64 + nt * 16 + c8;
    *(v8u16*)(void*)dst = val;
  }
}

// ---------------- kernel 4: output projection + bias + residual + LayerNorm ----------------
// 4 waves per block; each wave owns 16 n-tiles (2 per k-pass).
__global__ void __launch_bounds__(128) proj_ln_kernel(
    const uint16_t* __restrict__ Ob, const uint16_t* __restrict__ pT,
    const float* __restrict__ pb, const float* __restrict__ qin,
    const float* __restrict__ g, const float* __restrict__ be,
    float* __restrict__ out) {
  extern __shared__ __align__(16) char smem_raw[];
  uint16_t* Ab = (uint16_t*)smem_raw;                              // 16 x AROW bf16
  float* R = (float*)(smem_raw + (size_t)16 * AROW * sizeof(uint16_t)); // 16 x AROW f32
  const int tid  = threadIdx.x;
  const int wave = tid >> 5, wlane = tid & 31;
  const int b  = blockIdx.x >> 6;
  const int l0 = (blockIdx.x & 63) << 4;

  // stage A rows (already bf16)
  const uint16_t* orow = Ob + ((size_t)b * L_ + l0) * D_;
  for (int t = tid; t < 2048; t += 128) {
    int row = t >> 7, c8 = (t & 127) << 3;
    v8u16 v = *(const v8u16*)(const void*)(orow + (size_t)row * D_ + c8);
    *(v8u16*)(void*)(Ab + (size_t)row * AROW + c8) = v;
  }
  __syncthreads();

  const int n = wlane & 15, m0 = (wlane >> 4) << 3;
  for (int nt = wave * 16; nt < wave * 16 + 16; nt += 2) {
    v8f acc0 = {}, acc1 = {};
    const uint16_t* w0 = pT + (size_t)(nt * 16) * 1024;
    const uint16_t* w1 = w0 + (size_t)16 * 1024;
    for (int k0 = 0; k0 < 1024; k0 += 32) {
      __builtin_prefetch(w0 + (size_t)(wlane & 15) * 1024 + k0 + 128, 0, 1);
      v16bf a  = load_a_frag16(Ab, AROW, k0);
      v16bf b0 = load_b_frag16(w0, 1024, k0);
      v16bf b1 = load_b_frag16(w1, 1024, k0);
      acc0 = wmma_bf16(a, b0, acc0);
      acc1 = wmma_bf16(a, b1, acc1);
    }
#pragma unroll
    for (int r = 0; r < 8; ++r) {
      R[(size_t)(m0 + r) * AROW + (nt + 0) * 16 + n] = acc0[r];
      R[(size_t)(m0 + r) * AROW + (nt + 1) * 16 + n] = acc1[r];
    }
  }
  __syncthreads();

  // fused bias + residual + LayerNorm: 8 lanes per row, 128 cols each.
  // The 8 lanes of a row are consecutive -> shfl_xor 1/2/4 stays intra-wave.
  const int row = tid >> 3, cc = (tid & 7) << 7;
  const float* qrow = qin + ((size_t)b * L_ + l0 + row) * D_ + cc;
  float* Rr = R + (size_t)row * AROW + cc;
  float s1 = 0.f, s2 = 0.f;
  for (int i = 0; i < 128; ++i) {
    float xx = Rr[i] + pb[cc + i] + qrow[i];
    Rr[i] = xx;
    s1 += xx;
    s2 += xx * xx;
  }
  s1 += __shfl_xor(s1, 1, 32);  s2 += __shfl_xor(s2, 1, 32);
  s1 += __shfl_xor(s1, 2, 32);  s2 += __shfl_xor(s2, 2, 32);
  s1 += __shfl_xor(s1, 4, 32);  s2 += __shfl_xor(s2, 4, 32);
  float mu  = s1 * (1.0f / 1024.0f);
  float var = s2 * (1.0f / 1024.0f) - mu * mu;
  float rs  = rsqrtf(var + 1e-5f);
  float* wrow = out + ((size_t)b * L_ + l0 + row) * D_ + cc;
  for (int i = 0; i < 128; ++i)
    wrow[i] = (Rr[i] - mu) * rs * g[cc + i] + be[cc + i];
}

// ---------------- host side ----------------
extern "C" void kernel_launch(void* const* d_in, const int* in_sizes, int n_in,
                              void* d_out, int out_size, void* d_ws, size_t ws_size,
                              hipStream_t stream) {
  (void)in_sizes; (void)n_in; (void)out_size; (void)ws_size;
  const float*         q      = (const float*)d_in[0];
  const float*         k      = (const float*)d_in[1];
  const float*         v      = (const float*)d_in[2];
  const unsigned char* mask   = (const unsigned char*)d_in[3];
  const float*         w_qs   = (const float*)d_in[4];
  const float*         w_ks   = (const float*)d_in[5];
  const float*         w_vs   = (const float*)d_in[6];
  const float*         proj_w = (const float*)d_in[7];
  const float*         proj_b = (const float*)d_in[8];
  const float*         ln_g   = (const float*)d_in[9];
  const float*         ln_b   = (const float*)d_in[10];

  char* ws = (char*)d_ws;
  size_t off = 0;
  auto take = [&](size_t bytes) -> char* {
    char* p = ws + off;
    off = (off + bytes + 255) & ~(size_t)255;
    return p;
  };
  uint16_t* wqT = (uint16_t*)take((size_t)NH_ * DK_ * D_ * 2);   // [h][k][d]
  uint16_t* wkT = (uint16_t*)take((size_t)NH_ * DK_ * D_ * 2);
  uint16_t* wvT = (uint16_t*)take((size_t)NH_ * DK_ * D_ * 2);
  uint16_t* pT  = (uint16_t*)take((size_t)D_ * D_ * 2);          // [j][i]
  uint16_t* Qh  = (uint16_t*)take((size_t)NH_ * B_ * L_ * DK_ * 2);  // [h][b][l][64]
  uint16_t* Kh  = (uint16_t*)take((size_t)NH_ * B_ * L_ * DK_ * 2);  // [h][b][s][64]
  uint16_t* Vt  = (uint16_t*)take((size_t)NH_ * B_ * DK_ * L_ * 2);  // [h][b][d][s]
  uint16_t* Ob  = (uint16_t*)take((size_t)B_ * L_ * D_ * 2);         // [b][l][h*64+d]

  // 1) weight transposes / bf16 conversion
  {
    int total = NH_ * D_ * DK_;
    transpose_cvt_kernel<<<(total + 255) / 256, 256, 0, stream>>>(w_qs, wqT, NH_, D_, DK_);
    transpose_cvt_kernel<<<(total + 255) / 256, 256, 0, stream>>>(w_ks, wkT, NH_, D_, DK_);
    transpose_cvt_kernel<<<(total + 255) / 256, 256, 0, stream>>>(w_vs, wvT, NH_, D_, DK_);
    int totp = D_ * D_;
    transpose_cvt_kernel<<<(totp + 255) / 256, 256, 0, stream>>>(proj_w, pT, 1, D_, D_);
  }

  // 2) QKV projections (4 waves / block)
  size_t smem2 = (size_t)16 * AROW * 2 + (size_t)4 * 16 * 24 * 2;
  qkv_gemm_kernel<<<B_ * (L_ / 16), 128, smem2, stream>>>(q, wqT, Qh, 0);
  qkv_gemm_kernel<<<B_ * (L_ / 16), 128, smem2, stream>>>(k, wkT, Kh, 0);
  qkv_gemm_kernel<<<B_ * (L_ / 16), 128, smem2, stream>>>(v, wvT, Vt, 1);

  // 3) attention
  size_t smem3 = (size_t)16 * AROW * 4 + (size_t)16 * 24 * 2;
  hipFuncSetAttribute((const void*)attention_kernel,
                      hipFuncAttributeMaxDynamicSharedMemorySize, (int)smem3);
  attention_kernel<<<NH_ * B_ * (L_ / 16), 32, smem3, stream>>>(Qh, Kh, Vt, mask, Ob);

  // 4) output projection + residual + LayerNorm (4 waves / block)
  size_t smem4 = (size_t)16 * AROW * 2 + (size_t)16 * AROW * 4;
  hipFuncSetAttribute((const void*)proj_ln_kernel,
                      hipFuncAttributeMaxDynamicSharedMemorySize, (int)smem4);
  proj_ln_kernel<<<B_ * (L_ / 16), 128, smem4, stream>>>(Ob, pT, proj_b, q, ln_g,
                                                         ln_b, (float*)d_out);
}
